// LSTM_POS_Tagger_17721035063391
// MI455X (gfx1250) — matrix operand order repaired and checked by reading, not compiled, not run on hardware
//
#include <hip/hip_runtime.h>
#include <hip/hip_bf16.h>
#include <stdint.h>

// Problem sizes (from reference)
#define S_LEN 4096
#define L_CH  16
#define E_DIM 256
#define H_DIM 512
#define LBL   64
#define CH    4
#define KIN   260          // E + CH
#define KP    288          // KIN padded to multiple of 32 for 16x16x32 WMMA
#define G4H   (4*H_DIM)    // 2048 gate rows
#define NWG   16           // workgroups cooperating on the sequential word LSTM
#define SLICE (H_DIM/NWG)  // 32 hidden units per WG
#define ROWS  (4*SLICE)    // 128 gate rows per WG (i,f,g,o slices)
#define NB    4            // 16x16 N-tiles per wave in the GEMM (16x64 output)

typedef __attribute__((ext_vector_type(16))) __bf16 v16bf;
typedef __attribute__((ext_vector_type(8)))  float  v8f;

__device__ __forceinline__ float sigf(float x)      { return 1.0f/(1.0f+__expf(-x)); }
__device__ __forceinline__ float tanhfast(float x)  { return 2.0f*sigf(2.0f*x)-1.0f; }

// ---------------------------------------------------------------------------
// 0) init: zero the h ping-pong buffer and the grid-sync counter (ws is
//    poisoned 0xAA by the harness and not re-poisoned between replays).
// ---------------------------------------------------------------------------
__global__ void init_kernel(float* hbuf, unsigned* counter) {
  int i = blockIdx.x*blockDim.x + threadIdx.x;
  if (i < 2*H_DIM) hbuf[i] = 0.0f;
  if (i == 0) *counter = 0u;
}

// ---------------------------------------------------------------------------
// 1) char-level LSTM: one thread per word, CH=4 hidden, 16 masked steps.
// ---------------------------------------------------------------------------
__global__ __launch_bounds__(256)
void char_lstm_kernel(const float* __restrict__ chars, const int* __restrict__ lens,
                      const float* __restrict__ Wih,  const float* __restrict__ Whh,
                      const float* __restrict__ bih,  const float* __restrict__ bhh,
                      float* __restrict__ h_char) {
  int s = blockIdx.x*blockDim.x + threadIdx.x;
  if (s >= S_LEN) return;
  float wih[16], b[16], whh[16][4];
#pragma unroll
  for (int j = 0; j < 16; j++) {
    wih[j] = Wih[j]; b[j] = bih[j] + bhh[j];
#pragma unroll
    for (int u = 0; u < 4; u++) whh[j][u] = Whh[j*4+u];
  }
  float h[4] = {0,0,0,0}, c[4] = {0,0,0,0};
  int len = lens[s];
  for (int t = 0; t < L_CH; t++) {
    float x = chars[s*L_CH + t];
    float g[16];
#pragma unroll
    for (int j = 0; j < 16; j++) {
      float a = fmaf(x, wih[j], b[j]);
#pragma unroll
      for (int u = 0; u < 4; u++) a = fmaf(whh[j][u], h[u], a);
      g[j] = a;
    }
    if (t < len) {
#pragma unroll
      for (int u = 0; u < 4; u++) {
        float cn = sigf(g[4+u])*c[u] + sigf(g[u])*tanhfast(g[8+u]);
        h[u] = sigf(g[12+u])*tanhfast(cn);
        c[u] = cn;
      }
    }
  }
#pragma unroll
  for (int u = 0; u < 4; u++) h_char[s*CH + u] = h[u];
}

// ---------------------------------------------------------------------------
// 2) pack word_in = [emb[sentence] | h_char | 0-pad] -> bf16 [S, KP]
// ---------------------------------------------------------------------------
__global__ void pack_wordin_kernel(const int* __restrict__ sent, const float* __restrict__ emb,
                                   const float* __restrict__ h_char, __bf16* __restrict__ out) {
  int idx = blockIdx.x*blockDim.x + threadIdx.x;
  if (idx >= S_LEN*KP) return;
  int s = idx / KP, k = idx % KP;
  float v = 0.0f;
  if (k < E_DIM)    v = emb[(size_t)sent[s]*E_DIM + k];
  else if (k < KIN) v = h_char[s*CH + (k - E_DIM)];
  out[idx] = (__bf16)v;
}

// generic f32 [rows,cin] -> bf16 [rows,cout] with zero K-padding
__global__ void pack_rows_kernel(const float* __restrict__ src, __bf16* __restrict__ dst,
                                 int rows, int cin, int cout) {
  int idx = blockIdx.x*blockDim.x + threadIdx.x;
  if (idx >= rows*cout) return;
  int r = idx / cout, k = idx % cout;
  dst[idx] = (__bf16)(k < cin ? src[(size_t)r*cin + k] : 0.0f);
}

// ---------------------------------------------------------------------------
// 3) bf16 WMMA GEMM, register-blocked: each wave computes a 16x64 strip of
//    C[M,N] = A[M,Kp] * B[N,Kp]^T + bias[N] (f32 accumulate).
//    Per K-step: 1 A-fragment feeds NB=4 back-to-back v_wmma_f32_16x16x32_bf16
//    (2.5 b128 loads per WMMA instead of 4; A traffic cut 4x).
//    Fragment layouts per CDNA5 ISA 7.12.2:
//      A (16x32 bf16): lane m=lane&15; lanes<16 hold K[0..7],K[16..23],
//                      lanes>=16 hold K[8..15],K[24..31]
//      B (32x16 bf16): lane n=lane&15; lanes<16 hold K[0..7],K[16..23], lanes>=16 K[8..15],K[24..31]
//      C/D f32: VGPR r -> (M = (lane>>4)*8 + r, N = lane&15)
//    Requires N % 64 == 0 (N=2048 and N=64 here).
// ---------------------------------------------------------------------------
union Frag { uint4 q[2]; v16bf v; };

__global__ __launch_bounds__(128)
void wmma_gemm_bf16(const __bf16* __restrict__ A, const __bf16* __restrict__ B,
                    float* __restrict__ C, const float* __restrict__ bias,
                    int M, int N, int Kp) {
  int wave = (int)((blockIdx.x*blockDim.x + threadIdx.x) >> 5);
  int lane = threadIdx.x & 31;
  int ntc  = N >> 6;                       // 64-wide column chunks
  int tm = wave / ntc, tc = wave % ntc;
  if (tm >= (M >> 4)) return;
  int half = lane >> 4, l15 = lane & 15;
  const __bf16* arow  = A + (size_t)(tm*16 + l15) * Kp;
  const __bf16* brow0 = B + (size_t)(tc*64 + l15) * Kp;
  v8f acc[NB] = {};
  for (int k0 = 0; k0 < Kp; k0 += 32) {
    Frag a;
    a.q[0] = *(const uint4*)(arow + k0 + half*8);        // K k0..k0+7 / +8..15
    a.q[1] = *(const uint4*)(arow + k0 + half*8 + 16);   // K +16..23 / +24..31
#pragma unroll
    for (int nb = 0; nb < NB; nb++) {
      Frag bf;
      const uint4* bp = (const uint4*)(brow0 + (size_t)nb*16*Kp + k0 + half*8);
      bf.q[0] = bp[0];                                   // K k0..k0+7 / +8..15
      bf.q[1] = bp[2];                                   // K +16..23 / +24..31
      acc[nb] = __builtin_amdgcn_wmma_f32_16x16x32_bf16(false, a.v, false, bf.v,
                                                        (short)0, acc[nb], false, false);
    }
  }
#pragma unroll
  for (int nb = 0; nb < NB; nb++) {
    int col = tc*64 + nb*16 + l15;
    float bv = bias ? bias[col] : 0.0f;
#pragma unroll
    for (int r = 0; r < 8; r++) {
      int row = tm*16 + half*8 + r;
      C[(size_t)row*N + col] = acc[nb][r] + bv;
    }
  }
}

// ---------------------------------------------------------------------------
// 4) sequential word LSTM, hidden dim partitioned over NWG=16 WGPs.
//    Each WG keeps its 128x512 W_hh slice (256 KB) resident in LDS for all
//    4096 steps (async global->LDS DMA preload). h (512 f32) is exchanged
//    through L2 with a monotonic arrive/wait counter; h is double-buffered
//    so a single grid-sync per step suffices.
// ---------------------------------------------------------------------------
__global__ __launch_bounds__(256)
void word_lstm_kernel(const float* __restrict__ gx,    // [S, 2048] (incl. b_ih_w)
                      const float* __restrict__ Whh,   // [2048, 512]
                      const float* __restrict__ bhh,   // [2048]
                      float* __restrict__ hbuf,        // [2][512] ping-pong
                      unsigned* __restrict__ counter,
                      __bf16* __restrict__ hs) {       // [S, 512] bf16 out
  extern __shared__ char smem_raw[];
  float* lds_w    = (float*)smem_raw;          // [ROWS][H_DIM] = 256 KB
  float* h_sh     = lds_w + ROWS*H_DIM;        // [512]
  float* gates_sh = h_sh + H_DIM;              // [128]

  const int tid = threadIdx.x;
  const int kb  = blockIdx.x * SLICE;          // first hidden unit owned

  // --- async DMA preload: W_hh rows {g*512+kb+u} -> lds_w[g*32+u][:] ---
  {
    const int cpr = H_DIM/4;                   // 128 uint4 chunks per row
    for (int i = tid; i < ROWS*cpr; i += blockDim.x) {
      int r = i / cpr, cc = i % cpr;
      int g = r / SLICE, u = r % SLICE;
      const float* src = Whh + (size_t)(g*H_DIM + kb + u)*H_DIM + cc*4;
      unsigned lds_addr = (unsigned)(uintptr_t)(&lds_w[r*H_DIM + cc*4]);
      unsigned long long gaddr = (unsigned long long)(uintptr_t)src;
      asm volatile("global_load_async_to_lds_b128 %0, %1, off"
                   :: "v"(lds_addr), "v"(gaddr) : "memory");
    }
    asm volatile("s_wait_asynccnt 0x0" ::: "memory");
  }
  __syncthreads();

  // static per-thread work: 2 threads per gate row (256-elem half dots)
  const int r    = tid >> 1;                   // 0..127
  const int half = tid & 1;
  const int g_i  = r >> 5;                     // gate 0..3 (SLICE==32)
  const int u_i  = r & (SLICE-1);
  const int grow = g_i*H_DIM + kb + u_i;       // row in [0,2048)
  const float brow = bhh[grow];

  float c_state = 0.0f;                        // owned by tid < SLICE

  for (int t = 0; t < S_LEN; t++) {
    for (int i = tid; i < H_DIM; i += blockDim.x)
      h_sh[i] = hbuf[(t & 1)*H_DIM + i];
    __syncthreads();

    // 256-element partial dot from LDS (float4 = ds_load_b128)
    const float4* wv = (const float4*)&lds_w[r*H_DIM + half*(H_DIM/2)];
    const float4* hv = (const float4*)&h_sh[half*(H_DIM/2)];
    float acc = 0.0f;
#pragma unroll 8
    for (int k = 0; k < H_DIM/8; k++) {
      float4 w = wv[k], h4 = hv[k];
      acc = fmaf(w.x, h4.x, acc); acc = fmaf(w.y, h4.y, acc);
      acc = fmaf(w.z, h4.z, acc); acc = fmaf(w.w, h4.w, acc);
    }
    acc += __shfl_xor(acc, 1);                 // combine the two halves
    if (half == 0)
      gates_sh[r] = acc + gx[(size_t)t*G4H + grow] + brow;
    if (half == 0 && t + 1 < S_LEN)            // global_prefetch_b8 next gx row
      __builtin_prefetch(&gx[(size_t)(t+1)*G4H + grow], 0, 1);
    __syncthreads();

    if (tid < SLICE) {
      float gi = gates_sh[tid],           gf = gates_sh[SLICE + tid];
      float gg = gates_sh[2*SLICE + tid], go = gates_sh[3*SLICE + tid];
      c_state = sigf(gf)*c_state + sigf(gi)*tanhfast(gg);
      float hn = sigf(go)*tanhfast(c_state);
      hbuf[((t+1) & 1)*H_DIM + kb + tid] = hn;
      hs[(size_t)t*H_DIM + kb + tid] = (__bf16)hn;
      __threadfence();
    }
    __syncthreads();
    if (tid == 0) {                            // monotonic grid arrive/wait
      __hip_atomic_fetch_add(counter, 1u, __ATOMIC_ACQ_REL, __HIP_MEMORY_SCOPE_AGENT);
      const unsigned target = (unsigned)(NWG*(t+1));
      while (__hip_atomic_load(counter, __ATOMIC_ACQUIRE, __HIP_MEMORY_SCOPE_AGENT) < target)
        __builtin_amdgcn_s_sleep(1);
    }
    __syncthreads();
  }
}

// ---------------------------------------------------------------------------
// 5) row-wise log_softmax over 64 labels
// ---------------------------------------------------------------------------
__global__ __launch_bounds__(64)
void logsoftmax_kernel(const float* __restrict__ logits, float* __restrict__ out) {
  __shared__ float red[64];
  int row = blockIdx.x, tid = threadIdx.x;
  float v = logits[row*LBL + tid];
  red[tid] = v; __syncthreads();
  for (int s = 32; s > 0; s >>= 1) { if (tid < s) red[tid] = fmaxf(red[tid], red[tid+s]); __syncthreads(); }
  float m = red[0]; __syncthreads();
  red[tid] = __expf(v - m); __syncthreads();
  for (int s = 32; s > 0; s >>= 1) { if (tid < s) red[tid] += red[tid+s]; __syncthreads(); }
  out[row*LBL + tid] = v - (m + __logf(red[0]));
}

// ---------------------------------------------------------------------------
extern "C" void kernel_launch(void* const* d_in, const int* in_sizes, int n_in,
                              void* d_out, int out_size, void* d_ws, size_t ws_size,
                              hipStream_t stream) {
  const int*   sentence = (const int*)  d_in[0];
  const float* chars    = (const float*)d_in[1];
  const int*   lens     = (const int*)  d_in[2];
  const float* emb      = (const float*)d_in[3];
  const float* Wihc     = (const float*)d_in[4];
  const float* Whhc     = (const float*)d_in[5];
  const float* bihc     = (const float*)d_in[6];
  const float* bhhc     = (const float*)d_in[7];
  const float* Wihw     = (const float*)d_in[8];
  const float* Whhw     = (const float*)d_in[9];
  const float* bihw     = (const float*)d_in[10];
  const float* bhhw     = (const float*)d_in[11];
  const float* Wout     = (const float*)d_in[12];
  const float* bout     = (const float*)d_in[13];
  float* out = (float*)d_out;
  (void)in_sizes; (void)n_in; (void)out_size; (void)ws_size;

  char* ws = (char*)d_ws;
  size_t off = 0;
  auto carve = [&](size_t bytes) -> char* {
    char* p = ws + off; off = (off + bytes + 255) & ~(size_t)255; return p;
  };
  float*    h_char  = (float*)   carve((size_t)S_LEN*CH*4);
  __bf16*   word_in = (__bf16*)  carve((size_t)S_LEN*KP*2);
  __bf16*   Wihw_bf = (__bf16*)  carve((size_t)G4H*KP*2);
  float*    gx_w    = (float*)   carve((size_t)S_LEN*G4H*4);
  __bf16*   hs_bf   = (__bf16*)  carve((size_t)S_LEN*H_DIM*2);
  __bf16*   Wout_bf = (__bf16*)  carve((size_t)LBL*H_DIM*2);
  float*    logits  = (float*)   carve((size_t)S_LEN*LBL*4);
  float*    hbuf    = (float*)   carve((size_t)2*H_DIM*4);
  unsigned* counter = (unsigned*)carve(256);

  init_kernel<<<4, 256, 0, stream>>>(hbuf, counter);
  char_lstm_kernel<<<(S_LEN+255)/256, 256, 0, stream>>>(chars, lens, Wihc, Whhc, bihc, bhhc, h_char);
  pack_wordin_kernel<<<(S_LEN*KP+255)/256, 256, 0, stream>>>(sentence, emb, h_char, word_in);
  pack_rows_kernel<<<(G4H*KP+255)/256, 256, 0, stream>>>(Wihw, Wihw_bf, G4H, KIN, KP);
  pack_rows_kernel<<<(LBL*H_DIM+255)/256, 256, 0, stream>>>(Wout, Wout_bf, LBL, H_DIM, H_DIM);

  // gx_w[S, 2048] = word_in @ W_ih_w^T + b_ih_w   (bf16 WMMA, f32 acc)
  {
    int waves = (S_LEN/16) * (G4H/64);           // 16x64 strip per wave
    wmma_gemm_bf16<<<waves*32/128, 128, 0, stream>>>(word_in, Wihw_bf, gx_w, bihw,
                                                     S_LEN, G4H, KP);
  }

  // sequential scan: 16 WGPs, 256 KB LDS-resident W_hh slice each
  {
    size_t smem = (size_t)ROWS*H_DIM*4 + (size_t)H_DIM*4 + (size_t)ROWS*4;
    word_lstm_kernel<<<NWG, 256, smem, stream>>>(gx_w, Whhw, bhhw, hbuf, counter, hs_bf);
  }

  // logits[S, 64] = hs @ W_out^T + b_out   (bf16 WMMA, f32 acc)
  {
    int waves = (S_LEN/16) * (LBL/64);
    wmma_gemm_bf16<<<waves*32/128, 128, 0, stream>>>(hs_bf, Wout_bf, logits, bout,
                                                     S_LEN, LBL, H_DIM);
  }

  logsoftmax_kernel<<<S_LEN, 64, 0, stream>>>(logits, out);
}
